// SwinTransformerBlock_47605417508868
// MI455X (gfx1250) — matrix operand order repaired and checked by reading, not compiled, not run on hardware
//
#include <hip/hip_runtime.h>
#include <hip/hip_bf16.h>
#include <hip/hip_fp16.h>

// ---------------------------------------------------------------------------
// Swin block, MI455X (gfx1250, wave32, WMMA f32_16x16x32_f16 + TDM tile loads)
// B=8 H=W=112 C=256 NH=8 HD=32 WS=7 SS=3  N=49(->pad 64)  NW=256  HID=1024
// ---------------------------------------------------------------------------

typedef __attribute__((ext_vector_type(16))) _Float16 v16h;
typedef __attribute__((ext_vector_type(8)))  _Float16 v8h;
typedef __attribute__((ext_vector_type(8)))  float    v8f;
typedef __attribute__((ext_vector_type(4)))  unsigned int u32x4;
typedef __attribute__((ext_vector_type(8)))  int i32x8;
typedef __attribute__((ext_vector_type(4)))  int i32x4;

#define SS_ 3
#define SCALE_Q 0.17677669529663687f   // 32^-0.5

#if defined(__has_builtin)
#  if __has_builtin(__builtin_amdgcn_tensor_load_to_lds)
#    define HAVE_TDM 1
#  else
#    define HAVE_TDM 0
#  endif
#else
#  define HAVE_TDM 0
#endif

// ---- WMMA fragment helpers (layouts per CDNA5 ISA §7.12.2) -----

// A fragment 16x32 f16: lane l(0..15)=row l, K {0..7,16..23}; lane l+16: K {8..15,24..31}
__device__ __forceinline__ v16h load_afrag(const _Float16* p0, int stride) {
  int lane = threadIdx.x & 31;
  const _Float16* p = p0 + (size_t)(lane & 15) * stride + ((lane >> 4) << 3);
  v8h lo = *(const v8h*)p;
  v8h hi = *(const v8h*)(p + 16);
  v16h r;
#pragma unroll
  for (int i = 0; i < 8; ++i) { r[i] = lo[i]; r[i + 8] = hi[i]; }
  return r;
}

// B fragment 32x16 f16: lane l(0..15)=col l holds K {0..7,16..23}; lane l+16 holds K {8..15,24..31}.
// Source stored row-major [n][k].
__device__ __forceinline__ v16h load_bfrag(const _Float16* p0, int stride) {
  int lane = threadIdx.x & 31;
  const _Float16* p = p0 + (size_t)(lane & 15) * stride + ((lane >> 4) << 3);
  v8h lo = *(const v8h*)p;
  v8h hi = *(const v8h*)(p + 16);
  v16h r;
#pragma unroll
  for (int i = 0; i < 8; ++i) { r[i] = lo[i]; r[i + 8] = hi[i]; }
  return r;
}

__device__ __forceinline__ v8f wmma_f16(v16h a, v16h b, v8f c) {
  return __builtin_amdgcn_wmma_f32_16x16x32_f16(false, a, false, b, (short)0, c,
                                                false, false);
}

__device__ __forceinline__ float wred_sum32(float v) {
#pragma unroll
  for (int m = 16; m >= 1; m >>= 1) v += __shfl_xor(v, m, 32);
  return v;
}

#if HAVE_TDM
// Tensor Data Mover: DMA a [tile_m x tile_k] f16 tile (row stride K elems) from
// global into contiguous LDS. D# layout per CDNA5 ISA.
__device__ __forceinline__ void tdm_load_tile_f16(
    unsigned lds_byte_addr, const _Float16* gptr, int K, int tile_k, int tile_m) {
  unsigned long long ga = (unsigned long long)(size_t)gptr;
  u32x4 g0;
  g0[0] = 1u;                                                // count=1 (valid, user)
  g0[1] = lds_byte_addr;                                     // D#.lds_addr
  g0[2] = (unsigned)(ga & 0xffffffffull);                    // global_addr[31:0]
  g0[3] = (unsigned)((ga >> 32) & 0x01ffffffull) | (2u << 30); // ga[56:32] | type=2
  unsigned td0 = (unsigned)K;            // tensor_dim0 (elems)
  unsigned td1 = 0x40000000u;            // tensor_dim1: large, OOB never hit
  unsigned long long s0 = (unsigned long long)K;  // tensor_dim0_stride (elems)
  i32x8 g1;
  g1[0] = (int)(1u << 16);                                   // data_size=1 (2 bytes)
  g1[1] = (int)((td0 & 0xffffu) << 16);                      // tensor_dim0[15:0]
  g1[2] = (int)((td0 >> 16) | ((td1 & 0xffffu) << 16));      // td0[31:16], td1[15:0]
  g1[3] = (int)((td1 >> 16) | ((unsigned)tile_k << 16));     // td1[31:16], tile_dim0
  g1[4] = (int)(unsigned)tile_m;                             // tile_dim1 (tile_dim2=0)
  g1[5] = (int)(unsigned)(s0 & 0xffffffffull);               // stride0[31:0]
  g1[6] = (int)(unsigned)((s0 >> 32) & 0xffffull);           // stride0[47:32], stride1=0
  g1[7] = 0;
  i32x4 z4 = {0, 0, 0, 0};
#if __clang_major__ >= 23
  i32x8 z8 = {0, 0, 0, 0, 0, 0, 0, 0};
  __builtin_amdgcn_tensor_load_to_lds(g0, g1, z4, z4, z8, 0);
#else
  __builtin_amdgcn_tensor_load_to_lds(g0, g1, z4, z4, 0);
#endif
}
#endif

// ---------------------------------------------------------------------------
// Weight fp32 -> fp16 convert
__global__ __launch_bounds__(256) void f32_to_f16_kernel(
    const float* __restrict__ s, _Float16* __restrict__ d, int n) {
  int i = blockIdx.x * 256 + threadIdx.x;
  if (i < n) d[i] = (_Float16)s[i];
}

// ---------------------------------------------------------------------------
// LN1 + cyclic shift(-3,-3) + window partition -> Xw f16 [2048][64][256] (pad rows 0)
__global__ __launch_bounds__(256) void ln1_shift_window_kernel(
    const float* __restrict__ x, const float* __restrict__ g,
    const float* __restrict__ be, _Float16* __restrict__ Xw) {
  int w = blockIdx.x;
  int b = w >> 8, widx = w & 255;
  int wh = widx >> 4, ww = widx & 15;
  int wave = threadIdx.x >> 5, lane = threadIdx.x & 31;
#pragma unroll 1
  for (int rr = 0; rr < 8; ++rr) {
    int row = wave * 8 + rr;
    _Float16* dst = Xw + ((size_t)w * 64 + row) * 256;
    if (row < 49) {
      int th = row / 7, tw = row % 7;
      int sh = (wh * 7 + th + SS_) % 112;
      int sw = (ww * 7 + tw + SS_) % 112;
      const float* src = x + ((size_t)b * 12544 + sh * 112 + sw) * 256;
      float v[8]; float s = 0.f;
#pragma unroll
      for (int i = 0; i < 8; ++i) { v[i] = src[lane + i * 32]; s += v[i]; }
      float mean = wred_sum32(s) * (1.0f / 256.0f);
      float q = 0.f;
#pragma unroll
      for (int i = 0; i < 8; ++i) { float d2 = v[i] - mean; q += d2 * d2; }
      float rstd = rsqrtf(wred_sum32(q) * (1.0f / 256.0f) + 1e-5f);
#pragma unroll
      for (int i = 0; i < 8; ++i) {
        int c = lane + i * 32;
        dst[c] = (_Float16)((v[i] - mean) * rstd * g[c] + be[c]);
      }
    } else {
#pragma unroll
      for (int i = 0; i < 8; ++i) dst[lane + i * 32] = (_Float16)0.f;
    }
  }
}

// LN2: x1 fp32 [100352][256] -> h2 f16
__global__ __launch_bounds__(256) void ln2_kernel(
    const float* __restrict__ x1, const float* __restrict__ g,
    const float* __restrict__ be, _Float16* __restrict__ h2) {
  size_t row = (size_t)blockIdx.x * 8 + (threadIdx.x >> 5);
  int lane = threadIdx.x & 31;
  const float* src = x1 + row * 256;
  _Float16* dst = h2 + row * 256;
  float v[8]; float s = 0.f;
#pragma unroll
  for (int i = 0; i < 8; ++i) { v[i] = src[lane + i * 32]; s += v[i]; }
  float mean = wred_sum32(s) * (1.0f / 256.0f);
  float q = 0.f;
#pragma unroll
  for (int i = 0; i < 8; ++i) { float d2 = v[i] - mean; q += d2 * d2; }
  float rstd = rsqrtf(wred_sum32(q) * (1.0f / 256.0f) + 1e-5f);
#pragma unroll
  for (int i = 0; i < 8; ++i) {
    int c = lane + i * 32;
    dst[c] = (_Float16)((v[i] - mean) * rstd * g[c] + be[c]);
  }
}

// ---------------------------------------------------------------------------
// Tiled WMMA GEMM: C[M,N] = A[M,K] * Bw[N,K]^T (+epilogue)
// block = 128x128 tile, 8 waves (2 M x 4 N), wave = 64x32.
// A tile (128x32 f16) staged in LDS by the Tensor Data Mover (one wave issues
// tensor_load_to_lds, waits TENSORcnt, workgroup barrier releases consumers).
// MODE 0: QKV  -> scatter f16 to qkv buffer [win][3][head]  (q*SCALE, V transposed)
// MODE 1: MLP1 -> +bias, exact GELU, f16 out (stride 1024)
// MODE 2: MLP2 -> +bias +residual(x1), f32 out (stride 256)
template <int MODE>
__global__ __launch_bounds__(256) void gemm_kernel(
    const _Float16* __restrict__ A, const _Float16* __restrict__ Bw,
    const float* __restrict__ bias, const float* __restrict__ res,
    void* __restrict__ outp, int K) {
  __shared__ _Float16 Atile[128 * 32];
  int tid = threadIdx.x;
  int wid = tid >> 5, lane = tid & 31;
  int wm = wid & 1, wn = wid >> 1;
  int m0 = blockIdx.x * 128;
  int n0 = blockIdx.y * 128 + wn * 32;

  v8f acc[4][2];
  v8f zero = {0.f, 0.f, 0.f, 0.f, 0.f, 0.f, 0.f, 0.f};
#pragma unroll
  for (int i = 0; i < 4; ++i)
#pragma unroll
    for (int j = 0; j < 2; ++j) acc[i][j] = zero;

  const _Float16* aTileG = A + (size_t)m0 * K;
#if HAVE_TDM
  unsigned ldsAddr = (unsigned)(size_t)(void*)Atile;  // generic low 32b = LDS addr
#else
  int ldRow = tid >> 1;
  int ldCol = (tid & 1) << 4;
  const _Float16* aGlob = aTileG + (size_t)ldRow * K + ldCol;
  _Float16* aLds = Atile + ldRow * 32 + ldCol;
#endif

  for (int kk = 0; kk < K; kk += 32) {
    __syncthreads();
#if HAVE_TDM
    if (tid < 32) {   // single wave drives the TDM; EXEC is ignored by TDM ops
      tdm_load_tile_f16(ldsAddr, aTileG + kk, K, 32, 128);
      __builtin_amdgcn_s_wait_tensorcnt(0);
    }
#else
    v8h t0 = *(const v8h*)(aGlob + kk);
    v8h t1 = *(const v8h*)(aGlob + kk + 8);
    *(v8h*)aLds = t0;
    *(v8h*)(aLds + 8) = t1;
#endif
    __syncthreads();
    if (kk + 32 < K) __builtin_prefetch(aTileG + (size_t)kk + 32, 0, 1); // global_prefetch_b8
    v16h bf[2];
#pragma unroll
    for (int j = 0; j < 2; ++j)
      bf[j] = load_bfrag(Bw + (size_t)(n0 + j * 16) * K + kk, K);
#pragma unroll
    for (int i = 0; i < 4; ++i) {
      v16h af = load_afrag(Atile + (wm * 64 + i * 16) * 32, 32);
#pragma unroll
      for (int j = 0; j < 2; ++j) acc[i][j] = wmma_f16(af, bf[j], acc[i][j]);
    }
  }

  // C-fragment layout: VGPR r -> rows {r, r+8}, lanes 0..15 / 16..31, col = lane&15
  int colb = lane & 15;
  int rsel = (lane >> 4) * 8;
  int mW = m0 + wm * 64;
#pragma unroll
  for (int i = 0; i < 4; ++i)
#pragma unroll
    for (int j = 0; j < 2; ++j) {
      int col = n0 + j * 16 + colb;
#pragma unroll
      for (int r = 0; r < 8; ++r) {
        int row = mW + i * 16 + r + rsel;
        float v = acc[i][j][r] + bias[col];
        if (MODE == 0) {
          int m3 = col >> 8, head = (col >> 5) & 7, d = col & 31;
          if (m3 == 0) v *= SCALE_Q;
          int win = row >> 6, tok = row & 63;
          size_t base = (((size_t)win * 3 + m3) * 8 + head) * 2048;
          size_t dst = (m3 == 2) ? (base + (size_t)d * 64 + tok)     // V^T [d][tok]
                                 : (base + (size_t)tok * 32 + d);    // Q,K [tok][d]
          ((_Float16*)outp)[dst] = (_Float16)v;
        } else if (MODE == 1) {
          v = 0.5f * v * (1.0f + erff(v * 0.7071067811865475f));
          ((_Float16*)outp)[(size_t)row * 1024 + col] = (_Float16)v;
        } else {
          size_t idx = (size_t)row * 256 + col;
          ((float*)outp)[idx] = v + res[idx];
        }
      }
    }
}

// ---------------------------------------------------------------------------
// Per-window attention: 1 block = 1 window, 1 wave = 1 head.
// S = Q*K^T (1 WMMA k-step, HD=32) ; bias+mask+softmax in C-layout registers;
// P staged per-wave in LDS (64x64 f16) -> A-operand ; O = P*V via V^T frags.
__global__ __launch_bounds__(256) void attn_kernel(
    const _Float16* __restrict__ qkv, const float* __restrict__ rpb,
    _Float16* __restrict__ Ow) {
  extern __shared__ _Float16 smem[];   // 8 waves * 64*64 f16 = 64 KB
  int w = blockIdx.x;
  int h = threadIdx.x >> 5, lane = threadIdx.x & 31;
  _Float16* P = smem + h * 4096;
  const _Float16* Qb = qkv + (((size_t)w * 3 + 0) * 8 + h) * 2048;  // [64][32]
  const _Float16* Kb = qkv + (((size_t)w * 3 + 1) * 8 + h) * 2048;  // [64][32]
  const _Float16* Vt = qkv + (((size_t)w * 3 + 2) * 8 + h) * 2048;  // [32][64]

  v8f zero = {0.f, 0.f, 0.f, 0.f, 0.f, 0.f, 0.f, 0.f};
  v8f S[4][4];
  {
    v16h bf[4];
#pragma unroll
    for (int j = 0; j < 4; ++j) bf[j] = load_bfrag(Kb + j * 16 * 32, 32);
#pragma unroll
    for (int i = 0; i < 4; ++i) {
      v16h af = load_afrag(Qb + i * 16 * 32, 32);
#pragma unroll
      for (int j = 0; j < 4; ++j) S[i][j] = wmma_f16(af, bf[j], zero);
    }
  }

  int colb = lane & 15, rsel = (lane >> 4) * 8;
  int widx = w & 255, wh = widx >> 4, ww = widx & 15;
  int hh0 = wh * 7, ww0 = ww * 7;

  // column (key) coords / shift-region labels for this lane's 4 column tiles
  int mh[4], mw[4], mLab[4]; bool mValid[4];
#pragma unroll
  for (int j = 0; j < 4; ++j) {
    int m = j * 16 + colb;
    mValid[j] = (m < 49);
    int th = m / 7, tw = m % 7;
    mh[j] = th; mw[j] = tw;
    int gh = hh0 + th, gw = ww0 + tw;
    int rh = (gh < 105) ? 0 : ((gh < 109) ? 1 : 2);
    int rw = (gw < 105) ? 0 : ((gw < 109) ? 1 : 2);
    mLab[j] = rh * 3 + rw;
  }

#pragma unroll
  for (int i = 0; i < 4; ++i) {
#pragma unroll
    for (int r = 0; r < 8; ++r) {
      int n = i * 16 + r + rsel;     // query row (two rows per (i,r): one per lane-half)
      float sv[4];
      float mx = -3.0e38f;
      if (n < 49) {
        int nh = n / 7, nw = n % 7;
        int gh = hh0 + nh, gw = ww0 + nw;
        int rh = (gh < 105) ? 0 : ((gh < 109) ? 1 : 2);
        int rw = (gw < 105) ? 0 : ((gw < 109) ? 1 : 2);
        int nLab = rh * 3 + rw;
#pragma unroll
        for (int j = 0; j < 4; ++j) {
          float s;
          if (mValid[j]) {
            int dh = nh - mh[j] + 6, dw = nw - mw[j] + 6;
            s = S[i][j][r] + rpb[(dh * 13 + dw) * 8 + h];
            if (nLab != mLab[j]) s -= 100.0f;
          } else s = -3.0e38f;
          sv[j] = s;
          mx = fmaxf(mx, s);
        }
      } else {
#pragma unroll
        for (int j = 0; j < 4; ++j) sv[j] = -3.0e38f;
      }
#pragma unroll
      for (int t = 1; t < 16; t <<= 1) mx = fmaxf(mx, __shfl_xor(mx, t, 16));
      float sum = 0.f;
#pragma unroll
      for (int j = 0; j < 4; ++j) { float p = __expf(sv[j] - mx); sv[j] = p; sum += p; }
#pragma unroll
      for (int t = 1; t < 16; t <<= 1) sum += __shfl_xor(sum, t, 16);
      float inv = 1.0f / (sum + 1e-20f);
#pragma unroll
      for (int j = 0; j < 4; ++j)
        P[n * 64 + j * 16 + colb] = (_Float16)(sv[j] * inv);
    }
  }

  // O = P(64x64) * V(64x32): K = 64 -> two 16x16x32 steps; V as V^T B-fragments
  v8f O[4][2];
#pragma unroll
  for (int i = 0; i < 4; ++i)
#pragma unroll
    for (int jd = 0; jd < 2; ++jd) O[i][jd] = zero;
#pragma unroll
  for (int kt = 0; kt < 2; ++kt) {
    v16h bf[2];
#pragma unroll
    for (int jd = 0; jd < 2; ++jd)
      bf[jd] = load_bfrag(Vt + (size_t)(jd * 16) * 64 + kt * 32, 64);
#pragma unroll
    for (int i = 0; i < 4; ++i) {
      v16h af = load_afrag(P + (i * 16) * 64 + kt * 32, 64);
#pragma unroll
      for (int jd = 0; jd < 2; ++jd) O[i][jd] = wmma_f16(af, bf[jd], O[i][jd]);
    }
  }

  _Float16* dstW = Ow + (size_t)w * 64 * 256;
#pragma unroll
  for (int i = 0; i < 4; ++i)
#pragma unroll
    for (int jd = 0; jd < 2; ++jd) {
      int col = h * 32 + jd * 16 + colb;
#pragma unroll
      for (int r = 0; r < 8; ++r) {
        int tok = i * 16 + r + rsel;
        dstW[(size_t)tok * 256 + col] = (_Float16)O[i][jd][r];
      }
    }
}

// ---------------------------------------------------------------------------
// proj + window-reverse + unshift(+3,+3) + residual(x) -> x1 fp32
__global__ __launch_bounds__(256) void proj_residual_kernel(
    const _Float16* __restrict__ Ow, const _Float16* __restrict__ Wp,
    const float* __restrict__ pb, const float* __restrict__ x,
    float* __restrict__ x1) {
  int w = blockIdx.x;
  int wn = threadIdx.x >> 5, lane = threadIdx.x & 31;
  int n0 = wn * 32;
  const _Float16* Aw = Ow + (size_t)w * 64 * 256;

  v8f zero = {0.f, 0.f, 0.f, 0.f, 0.f, 0.f, 0.f, 0.f};
  v8f acc[4][2];
#pragma unroll
  for (int i = 0; i < 4; ++i)
#pragma unroll
    for (int j = 0; j < 2; ++j) acc[i][j] = zero;

#pragma unroll 2
  for (int kk = 0; kk < 256; kk += 32) {
    v16h bf[2];
#pragma unroll
    for (int j = 0; j < 2; ++j)
      bf[j] = load_bfrag(Wp + (size_t)(n0 + j * 16) * 256 + kk, 256);
#pragma unroll
    for (int i = 0; i < 4; ++i) {
      v16h af = load_afrag(Aw + (size_t)(i * 16) * 256 + kk, 256);
#pragma unroll
      for (int j = 0; j < 2; ++j) acc[i][j] = wmma_f16(af, bf[j], acc[i][j]);
    }
  }

  int colb = lane & 15, rsel = (lane >> 4) * 8;
  int b = w >> 8, widx = w & 255, wh = widx >> 4, ww = widx & 15;
#pragma unroll
  for (int i = 0; i < 4; ++i)
#pragma unroll
    for (int j = 0; j < 2; ++j) {
      int col = n0 + j * 16 + colb;
#pragma unroll
      for (int r = 0; r < 8; ++r) {
        int tok = i * 16 + r + rsel;
        if (tok < 49) {
          int th = tok / 7, tw = tok % 7;
          int gh = (wh * 7 + th + SS_) % 112;
          int gw = (ww * 7 + tw + SS_) % 112;
          size_t dst = ((size_t)b * 12544 + gh * 112 + gw) * 256 + col;
          x1[dst] = acc[i][j][r] + pb[col] + x[dst];
        }
      }
    }
}

// ---------------------------------------------------------------------------
extern "C" void kernel_launch(void* const* d_in, const int* in_sizes, int n_in,
                              void* d_out, int out_size, void* d_ws, size_t ws_size,
                              hipStream_t stream) {
  (void)in_sizes; (void)n_in; (void)out_size; (void)ws_size;
  const float* x      = (const float*)d_in[0];
  const float* n1g    = (const float*)d_in[1];
  const float* n1b    = (const float*)d_in[2];
  const float* qkv_w  = (const float*)d_in[3];
  const float* qkv_b  = (const float*)d_in[4];
  const float* rpb    = (const float*)d_in[5];
  const float* proj_w = (const float*)d_in[6];
  const float* proj_b = (const float*)d_in[7];
  const float* n2g    = (const float*)d_in[8];
  const float* n2b    = (const float*)d_in[9];
  const float* fc1_w  = (const float*)d_in[10];
  const float* fc1_b  = (const float*)d_in[11];
  const float* fc2_w  = (const float*)d_in[12];
  const float* fc2_b  = (const float*)d_in[13];

  char* ws = (char*)d_ws;
  // f16 weights
  _Float16* wq16 = (_Float16*)(ws + 0);          // 768*256
  _Float16* wp16 = (_Float16*)(ws + 393216);     // 256*256
  _Float16* wf1  = (_Float16*)(ws + 524288);     // 1024*256
  _Float16* wf2  = (_Float16*)(ws + 1048576);    // 256*1024
  // region A: Xw (LN1 windows, padded) then reused as h2 (LN2 out)
  size_t offA = (size_t)2 * 1024 * 1024;
  _Float16* Xw  = (_Float16*)(ws + offA);                    // 2048*64*256 f16 = 64 MB
  // region B: qkv buffer then reused as hid (MLP intermediate)
  size_t offB = offA + 67108864ull;
  _Float16* qkvb = (_Float16*)(ws + offB);                   // max(201.3, 205.5) MB
  // region C: attention output windows
  size_t offC = offB + 205520896ull;
  _Float16* Ow  = (_Float16*)(ws + offC);                    // 64 MB
  // region D: x1 residual (fp32)
  size_t offD = offC + 67108864ull;
  float* x1 = (float*)(ws + offD);                           // 100352*256 f32
  _Float16* h2  = Xw;
  _Float16* hid = qkvb;

  // 0) weights -> f16 (stay L2-resident: <1.6 MB total)
  f32_to_f16_kernel<<<(196608 + 255) / 256, 256, 0, stream>>>(qkv_w, wq16, 196608);
  f32_to_f16_kernel<<<(65536 + 255) / 256, 256, 0, stream>>>(proj_w, wp16, 65536);
  f32_to_f16_kernel<<<(262144 + 255) / 256, 256, 0, stream>>>(fc1_w, wf1, 262144);
  f32_to_f16_kernel<<<(262144 + 255) / 256, 256, 0, stream>>>(fc2_w, wf2, 262144);

  // 1) LN1 + shift + window partition (pad 49 -> 64 rows/window)
  ln1_shift_window_kernel<<<2048, 256, 0, stream>>>(x, n1g, n1b, Xw);
  // 2) QKV GEMM  [131072,256] x [256,768] -> scattered q,k,v^T f16
  gemm_kernel<0><<<dim3(1024, 6), 256, 0, stream>>>(Xw, wq16, qkv_b, nullptr,
                                                    (void*)qkvb, 256);
  // 3) windowed attention (bias + shift mask + softmax + P*V)
  attn_kernel<<<2048, 256, 65536, stream>>>(qkvb, rpb, Ow);
  // 4) proj + window reverse + unshift + residual -> x1 fp32
  proj_residual_kernel<<<2048, 256, 0, stream>>>(Ow, wp16, proj_b, x, x1);
  // 5) LN2
  ln2_kernel<<<12544, 256, 0, stream>>>(x1, n2g, n2b, h2);
  // 6) MLP fc1 + exact GELU   [100352,256] x [256,1024]
  gemm_kernel<1><<<dim3(784, 8), 256, 0, stream>>>(h2, wf1, fc1_b, nullptr,
                                                   (void*)hid, 256);
  // 7) MLP fc2 + residual -> d_out fp32   [100352,1024] x [1024,256]
  gemm_kernel<2><<<dim3(784, 2), 256, 0, stream>>>(hid, wf2, fc2_b, x1, d_out, 1024);
}
